// MAE_Net_21698174780229
// MI455X (gfx1250) — matrix operand order repaired and checked by reading, not compiled, not run on hardware
//
#include <hip/hip_runtime.h>
#include <math.h>

typedef __attribute__((ext_vector_type(16))) _Float16 v16h;
typedef __attribute__((ext_vector_type(8)))  _Float16 v8h;
typedef __attribute__((ext_vector_type(4)))  _Float16 f16x4;
typedef __attribute__((ext_vector_type(8)))  float    v8f;
typedef __attribute__((ext_vector_type(4)))  float    f32x4;

namespace {
constexpr int Bc = 8, Nn = 4096, Gc = 128, Kk = 16, Vv = 64, Mm = 64;
constexpr int Dd = 384, NHh = 8, FFf = 1536, DHh = 48;
constexpr int SDEC = 128;      // masked-decoder sequence length (M + V)
constexpr int LDSS = 40;       // generic kernel LDS row stride (halves)
constexpr int LSF  = 72;       // fast kernel LDS row stride (halves, 144B: 16B-aligned runs)
}

// =====================  WMMA GEMM, fast path  =====================
// Requires Kd % 64 == 0 and Mr % 32 == 0, B not transposed.
// Block = 4 waves, tile 32 rows x 64 cols; each wave owns two 16x16 accumulators.
// Register double-buffering: next tile's float4 loads overlap current WMMAs.
template<bool NFULL>
__global__ __launch_bounds__(128)
void gemm_wmma_fast_kernel(const float* __restrict__ A, const float* __restrict__ Bm,
                           const float* __restrict__ bias, const float* add, float* C,
                           int Nc, int Kd, int lda, int ldb, int ldc,
                           long long sAb, long long sAh, long long sBb, long long sBh,
                           long long sCb, long long sCh, int nh, float alpha, int doRelu)
{
    int z = blockIdx.z;
    int bb = z / nh, hd = z % nh;
    A  += (long long)bb * sAb + (long long)hd * sAh;
    Bm += (long long)bb * sBb + (long long)hd * sBh;
    long long coff = (long long)bb * sCb + (long long)hd * sCh;
    C += coff;
    const float* addp = add ? add + coff : nullptr;

    __shared__ __align__(16) _Float16 Als[32 * LSF];
    __shared__ __align__(16) _Float16 Bls[64 * LSF];

    int tid = threadIdx.x, wave = tid >> 5, lane = tid & 31;
    int row0 = blockIdx.y << 5, colB = blockIdx.x << 6;
    int r = lane & 15, h = lane >> 4;

    // A staging: 32x64 = 2048 f32 = 512 float4; 4 per thread (rows ar+8i)
    int ar = tid >> 4, ak = (tid & 15) * 4;
    const float* aP = A + (long long)(row0 + ar) * lda + ak;
    // B staging: 64x64 = 4096 f32 = 1024 float4; 8 per thread (k rows bk0+8i)
    int bk0 = tid >> 4, bn4 = (tid & 15) * 4;
    const float* bP = Bm + colB + bn4;
    const bool bOk = NFULL || (colB + bn4 + 4 <= Nc);

    f32x4 aR[4], bR[8];
    auto loadTiles = [&](int kb) {
#pragma unroll
        for (int i = 0; i < 4; ++i)
            aR[i] = *(const f32x4*)(aP + (long long)(i * 8) * lda + kb);
#pragma unroll
        for (int i = 0; i < 8; ++i) {
            f32x4 v = {0.f, 0.f, 0.f, 0.f};
            if (bOk) v = *(const f32x4*)(bP + (long long)(kb + bk0 + i * 8) * ldb);
            bR[i] = v;
        }
    };
    auto storeTiles = [&]() {
#pragma unroll
        for (int i = 0; i < 4; ++i)
            *(f16x4*)&Als[(ar + 8 * i) * LSF + ak] = __builtin_convertvector(aR[i], f16x4);
#pragma unroll
        for (int i = 0; i < 8; ++i) {
            f16x4 hv = __builtin_convertvector(bR[i], f16x4);
            int bk = bk0 + i * 8;          // transpose-store: Bls[n][k]
            Bls[(bn4 + 0) * LSF + bk] = hv[0];
            Bls[(bn4 + 1) * LSF + bk] = hv[1];
            Bls[(bn4 + 2) * LSF + bk] = hv[2];
            Bls[(bn4 + 3) * LSF + bk] = hv[3];
        }
    };

    v8f acc0 = {}, acc1 = {};
    loadTiles(0);
    for (int kb = 0; kb < Kd; kb += 64) {
        storeTiles();
        __syncthreads();
        if (kb + 64 < Kd) loadTiles(kb + 64);   // overlap with WMMAs below
#pragma unroll
        for (int kk = 0; kk < 64; kk += 32) {
            v8h a0lo = *(const v8h*)&Als[r * LSF + kk + h * 8];
            v8h a0hi = *(const v8h*)&Als[r * LSF + kk + 16 + h * 8];
            v16h af0 = __builtin_shufflevector(a0lo, a0hi, 0,1,2,3,4,5,6,7,8,9,10,11,12,13,14,15);
            v8h a1lo = *(const v8h*)&Als[(16 + r) * LSF + kk + h * 8];
            v8h a1hi = *(const v8h*)&Als[(16 + r) * LSF + kk + 16 + h * 8];
            v16h af1 = __builtin_shufflevector(a1lo, a1hi, 0,1,2,3,4,5,6,7,8,9,10,11,12,13,14,15);
            const _Float16* bp = &Bls[(wave * 16 + r) * LSF + kk + h * 8];
            v8h blo = *(const v8h*)bp;
            v8h bhi = *(const v8h*)(bp + 16);
            v16h bf = __builtin_shufflevector(blo, bhi, 0,1,2,3,4,5,6,7,8,9,10,11,12,13,14,15);
            acc0 = __builtin_amdgcn_wmma_f32_16x16x32_f16(false, af0, false, bf,
                                                          (short)0, acc0, false, false);
            acc1 = __builtin_amdgcn_wmma_f32_16x16x32_f16(false, af1, false, bf,
                                                          (short)0, acc1, false, false);
        }
        __syncthreads();
    }

    int col = colB + wave * 16 + r;
    if (NFULL || col < Nc) {
        float bval = bias ? bias[col] : 0.f;
#pragma unroll
        for (int i = 0; i < 8; ++i) {
            int row = row0 + i + h * 8;
            long long idx = (long long)row * ldc + col;
            float v = alpha * acc0[i] + bval;
            if (addp) v += addp[idx];
            if (doRelu) v = fmaxf(v, 0.f);
            C[idx] = v;
        }
#pragma unroll
        for (int i = 0; i < 8; ++i) {
            int row = row0 + 16 + i + h * 8;
            long long idx = (long long)row * ldc + col;
            float v = alpha * acc1[i] + bval;
            if (addp) v += addp[idx];
            if (doRelu) v = fmaxf(v, 0.f);
            C[idx] = v;
        }
    }
}

// =====================  WMMA GEMM, generic (odd K / transB)  =====================
// Branch-free staging: clamped addresses + value select (loads stay unconditional).
__global__ __launch_bounds__(128)
void gemm_wmma_kernel(const float* __restrict__ A, const float* __restrict__ Bm,
                      const float* __restrict__ bias, const float* add,
                      float* C, int Nc, int Kd,
                      int lda, int ldb, int ldc,
                      long long sAb, long long sAh, long long sBb, long long sBh,
                      long long sCb, long long sCh,
                      int nh, float alpha, int transB, int doRelu)
{
    int z = blockIdx.z;
    int bb = z / nh, hd = z % nh;
    A  += (long long)bb * sAb + (long long)hd * sAh;
    Bm += (long long)bb * sBb + (long long)hd * sBh;
    long long coff = (long long)bb * sCb + (long long)hd * sCh;
    C += coff;
    const float* addp = add ? add + coff : nullptr;

    __shared__ __align__(16) _Float16 Als[16 * LDSS];
    __shared__ __align__(16) _Float16 Bls[64 * LDSS];

    int tid  = threadIdx.x;
    int wave = tid >> 5, lane = tid & 31;
    int row0 = blockIdx.y << 4;
    int colB = blockIdx.x << 6;
    int r = lane & 15, h = lane >> 4;

    v8f acc = {};

    for (int kb = 0; kb < Kd; kb += 32) {
#pragma unroll
        for (int i = 0; i < 4; ++i) {          // A tile: 512 elems, 4 per thread
            int t = tid + i * 128;
            int ar = t >> 5, ak = t & 31;
            int gk = kb + ak;
            int cgk = (gk < Kd) ? gk : 0;      // clamp: load stays in bounds
            float v = A[(long long)(row0 + ar) * lda + cgk];
            Als[ar * LDSS + ak] = (_Float16)((gk < Kd) ? v : 0.f);
        }
#pragma unroll
        for (int i = 0; i < 16; ++i) {         // B tile: 2048 elems, 16 per thread
            int t = tid + i * 128;
            int bn = t >> 5, bk = t & 31;
            int gn = colB + bn, gk = kb + bk;
            bool ok = (gk < Kd) && (gn < Nc);
            int cn = ok ? gn : 0, ck = ok ? gk : 0;
            long long idx = transB ? (long long)cn * ldb + ck
                                   : (long long)ck * ldb + cn;
            float v = Bm[idx];
            Bls[bn * LDSS + bk] = (_Float16)(ok ? v : 0.f);
        }
        __syncthreads();
        v8h alo = *(const v8h*)&Als[r * LDSS + h * 8];
        v8h ahi = *(const v8h*)&Als[r * LDSS + 16 + h * 8];
        v16h af = __builtin_shufflevector(alo, ahi, 0,1,2,3,4,5,6,7,8,9,10,11,12,13,14,15);
        const _Float16* bp = &Bls[(wave * 16 + r) * LDSS + h * 8];
        v8h blo = *(const v8h*)bp;
        v8h bhi = *(const v8h*)(bp + 16);
        v16h bf = __builtin_shufflevector(blo, bhi, 0,1,2,3,4,5,6,7,8,9,10,11,12,13,14,15);
        acc = __builtin_amdgcn_wmma_f32_16x16x32_f16(false, af, false, bf,
                                                     (short)0, acc, false, false);
        __syncthreads();
    }

    int col = colB + wave * 16 + r;
    if (col < Nc) {
#pragma unroll
        for (int i = 0; i < 8; ++i) {
            int row = row0 + i + h * 8;
            long long idx = (long long)row * ldc + col;
            float v = alpha * acc[i];
            if (bias) v += bias[col];
            if (addp) v += addp[idx];
            if (doRelu) v = fmaxf(v, 0.f);
            C[idx] = v;
        }
    }
}

// =====================  elementwise / reduction kernels  =====================
__global__ __launch_bounds__(128)
void layernorm_kernel(const float* x, float* y, int rows, int D)
{
    int wid = blockIdx.x * 4 + (threadIdx.x >> 5);
    if (wid >= rows) return;
    int lane = threadIdx.x & 31;
    const float* xr = x + (long long)wid * D;
    float s = 0.f;
    for (int j = lane; j < D; j += 32) s += xr[j];
    for (int o = 16; o; o >>= 1) s += __shfl_xor(s, o, 32);
    float mean = s / D;
    float v = 0.f;
    for (int j = lane; j < D; j += 32) { float d = xr[j] - mean; v += d * d; }
    for (int o = 16; o; o >>= 1) v += __shfl_xor(v, o, 32);
    float inv = rsqrtf(v / D + 1e-5f);
    float* yr = y + (long long)wid * D;
    for (int j = lane; j < D; j += 32) yr[j] = (xr[j] - mean) * inv;
}

__global__ __launch_bounds__(128)
void softmax_mask_kernel(float* logits, const unsigned char* mask,
                         int B_, int NH_, int Sq, int Sk)
{
    long long row = (long long)blockIdx.x * 4 + (threadIdx.x >> 5);
    long long rows = (long long)B_ * NH_ * Sq;
    if (row >= rows) return;
    int lane = threadIdx.x & 31;
    int qi = (int)(row % Sq);
    int b  = (int)(row / ((long long)NH_ * Sq));
    float* lr = logits + row * Sk;
    const unsigned char* mr = mask ? mask + ((long long)b * Sq + qi) * Sk : nullptr;
    float vals[4];
    float mx = -3.4e38f;
    for (int j = lane; j < Sk; j += 32) {
        float v = lr[j];
        if (mr && !mr[j]) v = -1e9f;
        vals[(j - lane) >> 5] = v;
        mx = fmaxf(mx, v);
    }
    for (int o = 16; o; o >>= 1) mx = fmaxf(mx, __shfl_xor(mx, o, 32));
    float sum = 0.f;
    for (int j = lane; j < Sk; j += 32) {
        float e = __expf(vals[(j - lane) >> 5] - mx);
        vals[(j - lane) >> 5] = e; sum += e;
    }
    for (int o = 16; o; o >>= 1) sum += __shfl_xor(sum, o, 32);
    float inv = 1.f / sum;
    for (int j = lane; j < Sk; j += 32) lr[j] = vals[(j - lane) >> 5] * inv;
}

__global__ __launch_bounds__(256)
void fps_kernel(const float* pos, float* cen, int Np, int Gn)
{
    int b = blockIdx.x;
    const float* P = pos + (long long)b * Np * 3;
    float* Cn = cen + (long long)b * Gn * 3;
    __shared__ float mind[4096];
    __shared__ float rv[256];
    __shared__ int   ri[256];
    __shared__ float curp[3];
    int t = threadIdx.x;
    if (t == 0) {
        Cn[0] = P[0]; Cn[1] = P[1]; Cn[2] = P[2];
        curp[0] = P[0]; curp[1] = P[1]; curp[2] = P[2];
    }
    __syncthreads();
    for (int i = t; i < Np; i += 256) {
        float dx = P[3*i]-curp[0], dy = P[3*i+1]-curp[1], dz = P[3*i+2]-curp[2];
        mind[i] = dx*dx + dy*dy + dz*dz;
    }
    __syncthreads();
    for (int m = 1; m < Gn; ++m) {
        float bv = -1.f; int bi = 0x7fffffff;
        for (int i = t; i < Np; i += 256) {
            float v = mind[i];
            if (v > bv) { bv = v; bi = i; }
        }
        rv[t] = bv; ri[t] = bi; __syncthreads();
        for (int s = 128; s; s >>= 1) {
            if (t < s) {
                if (rv[t+s] > rv[t] || (rv[t+s] == rv[t] && ri[t+s] < ri[t])) {
                    rv[t] = rv[t+s]; ri[t] = ri[t+s];
                }
            }
            __syncthreads();
        }
        if (t == 0) {
            int nxt = ri[0];
            curp[0] = P[3*nxt]; curp[1] = P[3*nxt+1]; curp[2] = P[3*nxt+2];
            Cn[3*m] = curp[0]; Cn[3*m+1] = curp[1]; Cn[3*m+2] = curp[2];
        }
        __syncthreads();
        for (int i = t; i < Np; i += 256) {
            float dx = P[3*i]-curp[0], dy = P[3*i+1]-curp[1], dz = P[3*i+2]-curp[2];
            mind[i] = fminf(mind[i], dx*dx + dy*dy + dz*dz);
        }
        __syncthreads();
    }
}

__global__ void knn16_kernel(const float* q, const float* ref, int* out,
                             int B_, int Q, int R)
{
    long long tid = (long long)blockIdx.x * blockDim.x + threadIdx.x;
    if (tid >= (long long)B_ * Q) return;
    int b = (int)(tid / Q), qi = (int)(tid % Q);
    const float* qp = q + ((long long)b * Q + qi) * 3;
    float qx = qp[0], qy = qp[1], qz = qp[2];
    const float* rp = ref + (long long)b * R * 3;
    float bd[16]; int bi[16];
#pragma unroll
    for (int j = 0; j < 16; ++j) { bd[j] = 3.4e38f; bi[j] = 0; }
    for (int r = 0; r < R; ++r) {
        float dx = qx - rp[3*r], dy = qy - rp[3*r+1], dz = qz - rp[3*r+2];
        float d = dx*dx + dy*dy + dz*dz;
        if (d < bd[15]) {
            int p = 15;
            while (p > 0 && bd[p-1] > d) { bd[p] = bd[p-1]; bi[p] = bi[p-1]; --p; }
            bd[p] = d; bi[p] = r;
        }
    }
    int* o = out + ((long long)b * Q + qi) * 16;
#pragma unroll
    for (int j = 0; j < 16; ++j) o[j] = bi[j];
}

__global__ void group_rel_kernel(const float* pos, const float* cen, const int* nidx,
                                 float* rel, long long total)
{
    long long tid = (long long)blockIdx.x * blockDim.x + threadIdx.x;
    if (tid >= total) return;              // total = B*G*K
    long long bg = tid / Kk;
    int g = (int)(bg % Gc), b = (int)(bg / Gc);
    int pi = nidx[tid];
    const float* pp = pos + ((long long)b * Nn + pi) * 3;
    const float* cp = cen + ((long long)b * Gc + g) * 3;
    float* rr = rel + tid * 3;
    rr[0] = pp[0] - cp[0]; rr[1] = pp[1] - cp[1]; rr[2] = pp[2] - cp[2];
}

__global__ void max_over_k_kernel(const float* in, float* out, long long BG, int K_, int C)
{
    long long tid = (long long)blockIdx.x * blockDim.x + threadIdx.x;
    if (tid >= BG * C) return;
    long long bg = tid / C; int c = (int)(tid % C);
    float m = -3.4e38f;
    for (int k = 0; k < K_; ++k)
        m = fmaxf(m, in[(bg * K_ + k) * C + c]);
    out[bg * C + c] = m;
}

__global__ void build_cat_kernel(const float* gmax, const float* f, float* cat, long long total)
{
    long long tid = (long long)blockIdx.x * blockDim.x + threadIdx.x;
    if (tid >= total) return;              // total = B*G*K*512
    int c = (int)(tid % 512);
    long long bgk = tid / 512;
    long long bg = bgk / Kk;
    cat[tid] = (c < 256) ? gmax[bg * 256 + c] : f[bgk * 256 + (c - 256)];
}

__global__ void pe_kernel(const float* xyz, float* pe, long long total)
{
    long long tid = (long long)blockIdx.x * blockDim.x + threadIdx.x;
    if (tid >= total) return;              // total = rows * D
    int d = (int)(tid % Dd);
    long long bs = tid / Dd;
    int c = d / 128, jb = d % 128;
    float v = xyz[bs * 3 + c];
    float t = __powf(10000.f, (float)(jb & ~1) / 128.f);
    float p = v / t;
    pe[tid] = (jb & 1) ? __cosf(p) : __sinf(p);
}

__global__ void gather_rows_kernel(float* out, const float* in, const int* idx,
                                   int Q, int C, int Sin, long long total)
{
    long long tid = (long long)blockIdx.x * blockDim.x + threadIdx.x;
    if (tid >= total) return;              // total = B*Q*C
    int c = (int)(tid % C);
    long long bq = tid / C;
    int q = (int)(bq % Q), b = (int)(bq / Q);
    out[tid] = in[((long long)b * Sin + idx[(long long)b * Q + q]) * C + c];
}

__global__ void scatter_mask_kernel(unsigned char* mask, const int* nidx,
                                    int Q, int S, long long total)
{
    long long tid = (long long)blockIdx.x * blockDim.x + threadIdx.x;
    if (tid >= total) return;              // total = B*Q*16
    long long bq = tid / 16;
    mask[bq * S + nidx[tid]] = 1;
}

__global__ void add_kernel(float* dst, const float* a, const float* b, long long n)
{
    long long tid = (long long)blockIdx.x * blockDim.x + threadIdx.x;
    if (tid < n) dst[tid] = a[tid] + b[tid];
}

__global__ void build_q_kernel(float* out, const float* vis, const float* cpe,
                               const int* mskidx, const int* visidx, const float* mtok)
{
    long long tid = (long long)blockIdx.x * blockDim.x + threadIdx.x;
    long long total = (long long)Bc * SDEC * Dd;
    if (tid >= total) return;
    int d = (int)(tid % Dd);
    long long bi = tid / Dd;
    int i = (int)(bi % SDEC), b = (int)(bi / SDEC);
    float v, p;
    if (i < Mm) {
        v = mtok[d];
        p = cpe[((long long)b * Gc + mskidx[(long long)b * Mm + i]) * Dd + d];
    } else {
        v = vis[((long long)b * Vv + (i - Mm)) * Dd + d];
        p = cpe[((long long)b * Gc + visidx[(long long)b * Vv + (i - Mm)]) * Dd + d];
    }
    out[tid] = v + p;
}

__global__ void upsample_interp_kernel(const float* pos, const float* cen,
                                       const float* feats, float* xcat)
{
    long long tid = (long long)blockIdx.x * blockDim.x + threadIdx.x;
    if (tid >= (long long)Bc * Nn) return;
    int b = (int)(tid / Nn);
    const float* pp = pos + tid * 3;
    float px = pp[0], py = pp[1], pz = pp[2];
    const float* cb = cen + (long long)b * Gc * 3;
    float b0 = 3.4e38f, b1 = 3.4e38f, b2 = 3.4e38f;
    int i0 = 0, i1 = 0, i2 = 0;
    for (int g = 0; g < Gc; ++g) {
        float dx = px - cb[3*g], dy = py - cb[3*g+1], dz = pz - cb[3*g+2];
        float d = dx*dx + dy*dy + dz*dz;
        if (d < b0)      { b2=b1; i2=i1; b1=b0; i1=i0; b0=d; i0=g; }
        else if (d < b1) { b2=b1; i2=i1; b1=d;  i1=g; }
        else if (d < b2) { b2=d;  i2=g; }
    }
    float w0 = 1.f/(fmaxf(b0,0.f)+1e-8f), w1 = 1.f/(fmaxf(b1,0.f)+1e-8f),
          w2 = 1.f/(fmaxf(b2,0.f)+1e-8f);
    float s = w0 + w1 + w2; w0 /= s; w1 /= s; w2 /= s;
    const float* F  = feats + (long long)b * Gc * Dd;
    const float* F0 = F + (long long)i0 * Dd;
    const float* F1 = F + (long long)i1 * Dd;
    const float* F2 = F + (long long)i2 * Dd;
    float* o = xcat + tid * (Dd + 3);
    for (int c = 0; c < Dd; ++c)
        o[c] = w0 * F0[c] + w1 * F1[c] + w2 * F2[c];
    o[Dd] = px; o[Dd+1] = py; o[Dd+2] = pz;
}

// deterministic single-block MAE reduction
__global__ __launch_bounds__(256)
void mae_kernel(const float* decS, const float* decT,
                const float* outS, const float* outT,
                const int* mskS, const int* mskT, float* res)
{
    __shared__ float sh[256];
    int t = threadIdx.x;
    const long long tot = (long long)Bc * Mm * Dd;
    float acc = 0.f;
    for (long long i = t; i < tot; i += 256) {
        int d = (int)(i % Dd);
        long long bi = i / Dd;
        int ii = (int)(bi % Mm), b = (int)(bi / Mm);
        float tv = outS[((long long)b * Gc + mskS[(long long)b * Mm + ii]) * Dd + d];
        float pv = decS[((long long)b * SDEC + ii) * Dd + d];
        float a = fabsf(pv - tv);
        acc += (a < 2.f) ? 0.25f * a * a : (a - 1.f);
        tv = outT[((long long)b * Gc + mskT[(long long)b * Mm + ii]) * Dd + d];
        pv = decT[((long long)b * SDEC + ii) * Dd + d];
        a = fabsf(pv - tv);
        acc += (a < 2.f) ? 0.25f * a * a : (a - 1.f);
    }
    sh[t] = acc; __syncthreads();
    for (int s = 128; s; s >>= 1) { if (t < s) sh[t] += sh[t + s]; __syncthreads(); }
    if (t == 0) *res = sh[0] * (0.5f / (float)tot);
}

// =====================  host orchestration  =====================
static inline dim3 eg(long long n, int bs = 256) { return dim3((unsigned)((n + bs - 1) / bs)); }

extern "C" void kernel_launch(void* const* d_in, const int* in_sizes, int n_in,
                              void* d_out, int out_size, void* d_ws, size_t ws_size,
                              hipStream_t stream)
{
    (void)in_sizes; (void)n_in; (void)out_size; (void)ws_size;
    const float* pos_src = (const float*)d_in[0];
    const float* pos_tgt = (const float*)d_in[1];
    const int* vis_idx_s = (const int*)d_in[2];
    const int* msk_idx_s = (const int*)d_in[3];
    const int* vis_idx_t = (const int*)d_in[4];
    const int* msk_idx_t = (const int*)d_in[5];
    const float* tok_w1 = (const float*)d_in[6];
    const float* tok_b1 = (const float*)d_in[7];
    const float* tok_w2 = (const float*)d_in[8];
    const float* tok_b2 = (const float*)d_in[9];
    const float* tok_w3 = (const float*)d_in[10];
    const float* tok_b3 = (const float*)d_in[11];
    const float* tok_w4 = (const float*)d_in[12];
    const float* tok_b4 = (const float*)d_in[13];
    const float* mask_token = (const float*)d_in[14];
    const float* cx_attn = (const float*)d_in[15];
    const float* cx_ff1  = (const float*)d_in[16];
    const float* cx_ff2  = (const float*)d_in[17];
    const float* enc_sa  = (const float*)d_in[18];
    const float* enc_ca  = (const float*)d_in[19];
    const float* enc_ff1 = (const float*)d_in[20];
    const float* enc_ff2 = (const float*)d_in[21];
    const float* dec_ca  = (const float*)d_in[22];
    const float* dec_ff1 = (const float*)d_in[23];
    const float* dec_ff2 = (const float*)d_in[24];
    const float* up_w1 = (const float*)d_in[25];
    const float* up_b1 = (const float*)d_in[26];
    const float* up_w2 = (const float*)d_in[27];
    const float* up_b2 = (const float*)d_in[28];

    float* out = (float*)d_out;
    float* dense0 = out;
    float* dense1 = out + (long long)Bc * Nn * Dd;
    float* maeAcc = out + 2ll * Bc * Nn * Dd;

    // -------- workspace layout (bump allocated, same every call) --------
    float* W = (float*)d_ws;
    size_t o = 0;
    auto Al = [&](size_t n) { float* p = W + o; o += n; return p; };
    float* q_big = Al((size_t)Bc * Nn * Dd);
    float* lnq   = Al((size_t)Bc * Nn * Dd);
    float* ao    = Al((size_t)Bc * Nn * Dd);
    float* ffnh  = Al((size_t)Bc * Nn * FFf);          // also hosts tokenizer scratch
    float* probs = Al((size_t)Bc * NHh * Nn * Gc);     // also hosts xcat
    float* ppe   = Al((size_t)Bc * Nn * Dd);
    float* kbuf  = Al((size_t)Bc * Gc * Dd);
    float* vbuf  = Al((size_t)Bc * Gc * Dd);
    float* lnkv  = Al((size_t)Bc * Gc * Dd);
    float* lnm   = Al((size_t)Bc * Gc * Dd);
    float* mmbuf = Al((size_t)Bc * Gc * Dd);
    float* cenS = Al((size_t)Bc * Gc * 3);
    float* cenT = Al((size_t)Bc * Gc * 3);
    float* tokS = Al((size_t)Bc * Gc * Dd);   // becomes src_out
    float* tokT = Al((size_t)Bc * Gc * Dd);   // becomes tgt_out
    float* cpeS = Al((size_t)Bc * Gc * Dd);
    float* cpeT = Al((size_t)Bc * Gc * Dd);
    float* visS = Al((size_t)Bc * Vv * Dd);   // becomes vis_out_src
    float* visT = Al((size_t)Bc * Vv * Dd);
    float* vis3S = Al((size_t)Bc * Vv * 3);
    float* vis3T = Al((size_t)Bc * Vv * 3);
    float* decS = Al((size_t)Bc * SDEC * Dd);
    float* decT = Al((size_t)Bc * SDEC * Dd);
    int* nidx = (int*)Al((size_t)Bc * Gc * 16);
    unsigned char* mfS = (unsigned char*)Al((size_t)Bc * Gc * Gc / 4);
    unsigned char* mfT = (unsigned char*)Al((size_t)Bc * Gc * Gc / 4);
    unsigned char* mvS = (unsigned char*)Al((size_t)Bc * Vv * Vv / 4);
    unsigned char* mvT = (unsigned char*)Al((size_t)Bc * Vv * Vv / 4);

    // tokenizer scratch aliases the big FFN buffer (dead at that time)
    float* rel  = ffnh;
    float* f1b  = rel  + (size_t)Bc * Gc * Kk * 3;
    float* f2b  = f1b  + (size_t)Bc * Gc * Kk * 128;
    float* gmaxb= f2b  + (size_t)Bc * Gc * Kk * 256;
    float* catb = gmaxb+ (size_t)Bc * Gc * 256;
    float* hb   = catb + (size_t)Bc * Gc * Kk * 512;
    float* tpb  = hb   + (size_t)Bc * Gc * Kk * 512;
    float* xcat = probs;                                // upsample concat buffer

    // -------- helpers --------
    auto gemm = [&](const float* Ap, const float* Bp, const float* bias, const float* add,
                    float* Cp, int Mr, int Nc, int Kd, int lda, int ldb, int ldc,
                    long long sAb, long long sAh, long long sBb, long long sBh,
                    long long sCb, long long sCh, int nb, int nh,
                    float alpha, bool tB, bool relu) {
        if (!tB && (Kd % 64 == 0) && (Mr % 32 == 0)) {
            dim3 g((Nc + 63) / 64, Mr / 32, nb * nh);
            if (Nc % 64 == 0)
                gemm_wmma_fast_kernel<true><<<g, 128, 0, stream>>>(Ap, Bp, bias, add, Cp,
                    Nc, Kd, lda, ldb, ldc, sAb, sAh, sBb, sBh, sCb, sCh, nh, alpha, relu ? 1 : 0);
            else
                gemm_wmma_fast_kernel<false><<<g, 128, 0, stream>>>(Ap, Bp, bias, add, Cp,
                    Nc, Kd, lda, ldb, ldc, sAb, sAh, sBb, sBh, sCb, sCh, nh, alpha, relu ? 1 : 0);
        } else {
            dim3 g((Nc + 63) / 64, Mr / 16, nb * nh);
            gemm_wmma_kernel<<<g, 128, 0, stream>>>(Ap, Bp, bias, add, Cp, Nc, Kd,
                lda, ldb, ldc, sAb, sAh, sBb, sBh, sCb, sCh, nh, alpha, tB ? 1 : 0, relu ? 1 : 0);
        }
    };
    auto gemm_flat = [&](const float* Ap, int rows, int Kd, int Nc, const float* Wp,
                         const float* bias, const float* add, float* Cp, bool relu) {
        gemm(Ap, Wp, bias, add, Cp, rows, Nc, Kd, Kd, Nc, Nc,
             0, 0, 0, 0, 0, 0, 1, 1, 1.0f, false, relu);
    };
    auto ln = [&](const float* x, float* y, int rows) {
        layernorm_kernel<<<dim3((rows + 3) / 4), 128, 0, stream>>>(x, y, rows, Dd);
    };
    auto run_attn = [&](const float* lnq_p, int Sq, const float* lnkv_p, int Sk,
                        const float* w4, const unsigned char* mask,
                        float* outp, const float* resp) {
        gemm_flat(lnq_p,  Bc * Sq, Dd, Dd, w4,              nullptr, nullptr, q_big, false);
        gemm_flat(lnkv_p, Bc * Sk, Dd, Dd, w4 + Dd * Dd,    nullptr, nullptr, kbuf,  false);
        gemm_flat(lnkv_p, Bc * Sk, Dd, Dd, w4 + 2 * Dd * Dd,nullptr, nullptr, vbuf,  false);
        // logits = (1/sqrt(DH)) * Q x K^T  per (b,h)  [Kd=48 -> generic transB path]
        gemm(q_big, kbuf, nullptr, nullptr, probs, Sq, Sk, DHh, Dd, Dd, Sk,
             (long long)Sq * Dd, DHh, (long long)Sk * Dd, DHh,
             (long long)NHh * Sq * Sk, (long long)Sq * Sk,
             Bc, NHh, 0.14433756729740643f, true, false);
        long long rows = (long long)Bc * NHh * Sq;
        softmax_mask_kernel<<<dim3((unsigned)((rows + 3) / 4)), 128, 0, stream>>>(
            probs, mask, Bc, NHh, Sq, Sk);
        // out = P x V per (b,h)   [Kd=Sk aligned, Nc=48 -> fast<NFULL=false>]
        gemm(probs, vbuf, nullptr, nullptr, ao, Sq, DHh, Sk, Sk, Dd, Dd,
             (long long)NHh * Sq * Sk, (long long)Sq * Sk, (long long)Sk * Dd, DHh,
             (long long)Sq * Dd, DHh, Bc, NHh, 1.0f, false, false);
        gemm_flat(ao, Bc * Sq, Dd, Dd, w4 + 3 * Dd * Dd, nullptr, resp, outp, false);
    };
    auto ffn = [&](float* x, int S, const float* w1, const float* w2) {
        ln(x, lnq, Bc * S);
        gemm_flat(lnq, Bc * S, Dd, FFf, w1, nullptr, nullptr, ffnh, true);
        gemm_flat(ffnh, Bc * S, FFf, Dd, w2, nullptr, x, x, false);
    };
    auto cross_enc = [&](float* sb, float* tb, int S,
                         const unsigned char* ms, const unsigned char* mt) {
        ln(sb, lnq, Bc * S); run_attn(lnq, S, lnq, S, cx_attn, ms, sb, sb);
        ln(tb, lnq, Bc * S); run_attn(lnq, S, lnq, S, cx_attn, mt, tb, tb);
        ln(sb, lnq, Bc * S); ln(tb, lnkv, Bc * S);
        run_attn(lnq,  S, lnkv, S, cx_attn + 4 * Dd * Dd, nullptr, sb, sb);
        run_attn(lnkv, S, lnq,  S, cx_attn + 4 * Dd * Dd, nullptr, tb, tb);
        ffn(sb, S, cx_ff1, cx_ff2);
        ffn(tb, S, cx_ff1, cx_ff2);
    };
    auto dec_run = [&](float* x, int Sq, const float* mem, const float* mempe,
                       const float* sa, const float* ca,
                       const float* f1, const float* f2, int layers) {
        long long nm = (long long)Bc * Gc * Dd;
        add_kernel<<<eg(nm), 256, 0, stream>>>(mmbuf, mem, mempe, nm);
        ln(mmbuf, lnm, Bc * Gc);
        for (int l = 0; l < layers; ++l) {
            if (sa) {
                ln(x, lnq, Bc * Sq);
                run_attn(lnq, Sq, lnq, Sq, sa + (long long)l * 4 * Dd * Dd, nullptr, x, x);
            }
            ln(x, lnq, Bc * Sq);
            run_attn(lnq, Sq, lnm, Gc, ca + (long long)l * 4 * Dd * Dd, nullptr, x, x);
            ffn(x, Sq, f1 + (long long)l * Dd * FFf, f2 + (long long)l * FFf * Dd);
        }
    };

    // -------- 1) tokenize both clouds --------
    for (int side = 0; side < 2; ++side) {
        const float* pos = side ? pos_tgt : pos_src;
        float* cen = side ? cenT : cenS;
        float* tok = side ? tokT : tokS;
        fps_kernel<<<Bc, 256, 0, stream>>>(pos, cen, Nn, Gc);
        knn16_kernel<<<eg((long long)Bc * Gc, 128), 128, 0, stream>>>(cen, pos, nidx, Bc, Gc, Nn);
        group_rel_kernel<<<eg((long long)Bc * Gc * Kk), 256, 0, stream>>>(pos, cen, nidx, rel,
                                                                          (long long)Bc * Gc * Kk);
        gemm_flat(rel, Bc * Gc * Kk, 3, 128, tok_w1, tok_b1, nullptr, f1b, true);
        gemm_flat(f1b, Bc * Gc * Kk, 128, 256, tok_w2, tok_b2, nullptr, f2b, false);
        max_over_k_kernel<<<eg((long long)Bc * Gc * 256), 256, 0, stream>>>(f2b, gmaxb,
                                                                (long long)Bc * Gc, Kk, 256);
        build_cat_kernel<<<eg((long long)Bc * Gc * Kk * 512), 256, 0, stream>>>(gmaxb, f2b, catb,
                                                                (long long)Bc * Gc * Kk * 512);
        gemm_flat(catb, Bc * Gc * Kk, 512, 512, tok_w3, tok_b3, nullptr, hb, true);
        gemm_flat(hb, Bc * Gc * Kk, 512, Dd, tok_w4, tok_b4, nullptr, tpb, false);
        max_over_k_kernel<<<eg((long long)Bc * Gc * Dd), 256, 0, stream>>>(tpb, tok,
                                                                (long long)Bc * Gc, Kk, Dd);
    }

    // -------- 2) center PE, gathers, neighborhood masks --------
    pe_kernel<<<eg((long long)Bc * Gc * Dd), 256, 0, stream>>>(cenS, cpeS, (long long)Bc * Gc * Dd);
    pe_kernel<<<eg((long long)Bc * Gc * Dd), 256, 0, stream>>>(cenT, cpeT, (long long)Bc * Gc * Dd);
    gather_rows_kernel<<<eg((long long)Bc * Vv * 3), 256, 0, stream>>>(vis3S, cenS, vis_idx_s, Vv, 3, Gc, (long long)Bc * Vv * 3);
    gather_rows_kernel<<<eg((long long)Bc * Vv * 3), 256, 0, stream>>>(vis3T, cenT, vis_idx_t, Vv, 3, Gc, (long long)Bc * Vv * 3);
    gather_rows_kernel<<<eg((long long)Bc * Vv * Dd), 256, 0, stream>>>(visS, tokS, vis_idx_s, Vv, Dd, Gc, (long long)Bc * Vv * Dd);
    gather_rows_kernel<<<eg((long long)Bc * Vv * Dd), 256, 0, stream>>>(visT, tokT, vis_idx_t, Vv, Dd, Gc, (long long)Bc * Vv * Dd);

    hipMemsetAsync(mfS, 0, (size_t)Bc * Gc * Gc, stream);
    knn16_kernel<<<eg((long long)Bc * Gc, 128), 128, 0, stream>>>(cenS, cenS, nidx, Bc, Gc, Gc);
    scatter_mask_kernel<<<eg((long long)Bc * Gc * 16), 256, 0, stream>>>(mfS, nidx, Gc, Gc, (long long)Bc * Gc * 16);
    hipMemsetAsync(mfT, 0, (size_t)Bc * Gc * Gc, stream);
    knn16_kernel<<<eg((long long)Bc * Gc, 128), 128, 0, stream>>>(cenT, cenT, nidx, Bc, Gc, Gc);
    scatter_mask_kernel<<<eg((long long)Bc * Gc * 16), 256, 0, stream>>>(mfT, nidx, Gc, Gc, (long long)Bc * Gc * 16);
    hipMemsetAsync(mvS, 0, (size_t)Bc * Vv * Vv, stream);
    knn16_kernel<<<eg((long long)Bc * Vv, 128), 128, 0, stream>>>(vis3S, vis3S, nidx, Bc, Vv, Vv);
    scatter_mask_kernel<<<eg((long long)Bc * Vv * 16), 256, 0, stream>>>(mvS, nidx, Vv, Vv, (long long)Bc * Vv * 16);
    hipMemsetAsync(mvT, 0, (size_t)Bc * Vv * Vv, stream);
    knn16_kernel<<<eg((long long)Bc * Vv, 128), 128, 0, stream>>>(vis3T, vis3T, nidx, Bc, Vv, Vv);
    scatter_mask_kernel<<<eg((long long)Bc * Vv * 16), 256, 0, stream>>>(mvT, nidx, Vv, Vv, (long long)Bc * Vv * 16);

    // -------- 3) cross encoders (vis stream, then full stream) --------
    cross_enc(visS, visT, Vv, mvS, mvT);   // -> vis_out_{src,tgt}
    cross_enc(tokS, tokT, Gc, mfS, mfT);   // -> src_out / tgt_out

    // -------- 4) masked decoders --------
    build_q_kernel<<<eg((long long)Bc * SDEC * Dd), 256, 0, stream>>>(decS, visS, cpeS, msk_idx_s, vis_idx_s, mask_token);
    build_q_kernel<<<eg((long long)Bc * SDEC * Dd), 256, 0, stream>>>(decT, visT, cpeT, msk_idx_t, vis_idx_t, mask_token);
    dec_run(decS, SDEC, tokT, cpeT, enc_sa, enc_ca, enc_ff1, enc_ff2, 4);
    dec_run(decT, SDEC, tokS, cpeS, enc_sa, enc_ca, enc_ff1, enc_ff2, 4);

    // -------- 5) MAE loss (deterministic single-block reduction) --------
    mae_kernel<<<1, 256, 0, stream>>>(decS, decT, tokS, tokT, msk_idx_s, msk_idx_t, maeAcc);

    // -------- 6) upsample + dense decoders (dense state lives in d_out) --------
    for (int side = 0; side < 2; ++side) {
        const float* pos = side ? pos_tgt : pos_src;
        const float* cen = side ? cenT : cenS;
        const float* feats = side ? tokT : tokS;
        const float* cpe = side ? cpeT : cpeS;
        float* xst = side ? dense1 : dense0;
        pe_kernel<<<eg((long long)Bc * Nn * Dd), 256, 0, stream>>>(pos, ppe, (long long)Bc * Nn * Dd);
        upsample_interp_kernel<<<eg((long long)Bc * Nn), 256, 0, stream>>>(pos, cen, feats, xcat);
        gemm_flat(xcat, Bc * Nn, Dd + 3, Dd, up_w1, up_b1, nullptr, q_big, true);
        gemm_flat(q_big, Bc * Nn, Dd, Dd, up_w2, up_b2, nullptr, xst, true);
        add_kernel<<<eg((long long)Bc * Nn * Dd), 256, 0, stream>>>(xst, xst, ppe, (long long)Bc * Nn * Dd);
        dec_run(xst, Nn, feats, cpe, nullptr, dec_ca, dec_ff1, dec_ff2, 2);
    }
}
